// AnotherMamba_45165876084738
// MI455X (gfx1250) — matrix operand order repaired and checked by reading, not compiled, not run on hardware
//
#include <hip/hip_runtime.h>
#include <cstdint>
#include <cstddef>

// Problem constants (match reference)
#define B_SZ  2
#define L_SZ  4096
#define D_M   2048
#define D_IN  4096
#define D_XB  1024
#define D_S   16
#define D_C   4
#define D_TR  128

typedef __bf16 bf16_t;
typedef bf16_t v16bf __attribute__((ext_vector_type(16)));
typedef float  v8f   __attribute__((ext_vector_type(8)));
typedef unsigned int v4u __attribute__((ext_vector_type(4)));
typedef int v8i __attribute__((ext_vector_type(8)));
typedef int v4i __attribute__((ext_vector_type(4)));

static __device__ __forceinline__ bf16_t f2bf(float f) {
  // round-to-nearest-even fp32 -> bf16
  uint32_t u = __builtin_bit_cast(uint32_t, f);
  u += 0x7FFFu + ((u >> 16) & 1u);
  uint16_t h = (uint16_t)(u >> 16);
  return __builtin_bit_cast(bf16_t, h);
}

// ---------------------------------------------------------------------------
// TDM: issue a 2D bf16 tile DMA global -> LDS.
// D# group0/group1 packed per CDNA5 ISA (CDNA5 ISA):
//   g0: count=1 | lds_addr | global_addr(57b) | type=2
//   g1: data_size=1(2B), pad_enable, pad_interval=3 (16 DWORDs = one 64B row),
//       pad_amount=3 (4 DWORDs = 16B)  ->  LDS row stride 80B, bank-friendly
//   tensor dims = remaining extents from tile start (OOB returns zeros),
//   tile = tile_k x tile_rows, row stride = K elements.
// Groups 2/3 (and the trailing group of the 6-arg builtin) zero: 2D tensor.
// Tracked by TENSORcnt.
// ---------------------------------------------------------------------------
static __device__ __forceinline__ void tdm_load_tile_bf16(
    unsigned lds_off, const bf16_t* gsrc, unsigned rem_k, unsigned rem_rows,
    unsigned tile_k, unsigned tile_rows, unsigned long long row_stride)
{
  const unsigned long long ga = (unsigned long long)(uintptr_t)gsrc;
  v4u g0 = { 1u,                                   // count=1, user mode
             lds_off,                              // LDS byte address
             (unsigned)ga,                         // global addr [31:0]
             (unsigned)((ga >> 32) & 0x01FFFFFFull) | (2u << 30) }; // [56:32]|type=2
  v8i g1;
  g1[0] = (int)((1u << 16)      // data_size = 1 -> 2 bytes/elem
              | (1u << 20)      // pad_enable
              | (3u << 22)      // pad_interval code 3 -> 16 DWORDs
              | (3u << 25));    // pad_amount  code 3 -> 4 DWORDs (16B)
  g1[1] = (int)((rem_k & 0xFFFFu) << 16);                              // dim0[15:0]
  g1[2] = (int)(((rem_k >> 16) & 0xFFFFu) | ((rem_rows & 0xFFFFu) << 16)); // dim0[31:16]|dim1[15:0]
  g1[3] = (int)(((rem_rows >> 16) & 0xFFFFu) | (tile_k << 16));        // dim1[31:16]|tile_dim0
  g1[4] = (int)(tile_rows & 0xFFFFu);                                  // tile_dim1 (tile_dim2=0)
  g1[5] = (int)(row_stride & 0xFFFFFFFFull);                           // dim0_stride[31:0]
  g1[6] = (int)((row_stride >> 32) & 0xFFFFull);                       // dim0_stride[47:32]
  g1[7] = 0;
  const v4i z4 = {0, 0, 0, 0};
  const v8i z8 = {0, 0, 0, 0, 0, 0, 0, 0};
  __builtin_amdgcn_tensor_load_to_lds(g0, g1, z4, z4, z8, 0);
}

// ---------------------------------------------------------------------------
// GEMM: O[M,N] = A[M,K](bf16) * W[N,K](bf16)^T (+ bias[N]), fp32 accumulate.
// BM=BN=128, BK=32, 256 threads = 8 wave32 (4 m-waves x 2 n-waves),
// each wave computes 32x64 via 2x4 v_wmma_f32_16x16x32_bf16 tiles.
// Global->LDS via TDM (wave 0 issues, double-buffered one tile ahead);
// all waves only do ds_load_b128 + wmma in the steady state.
// Requires M%128==0, N%128==0, K%32==0 (true for every call here).
// ---------------------------------------------------------------------------
#define BM 128
#define BN 128
#define BK 32
#define LDK 40   // 32 bf16 (64B) + TDM pad 16B = 80B row stride

__global__ __launch_bounds__(256) void gemm_bf16_tdm(
    const bf16_t* __restrict__ A, const bf16_t* __restrict__ W,
    const float* __restrict__ bias, float* __restrict__ O,
    int M, int N, int K)
{
  __shared__ bf16_t sA[2][BM][LDK];
  __shared__ bf16_t sB[2][BN][LDK];

  const int tid  = threadIdx.x;
  const int lane = tid & 31;
  const int wave = tid >> 5;
  const int hi   = lane >> 4;   // half-wave index (0/1)
  const int ln   = lane & 15;
  const int wm   = wave & 3;    // 4 waves along M
  const int wn   = wave >> 2;   // 2 waves along N
  const int bm   = blockIdx.y * BM;
  const int bn   = blockIdx.x * BN;

  v8f acc[2][4] = {};
  const int nk = K / BK;

  auto issue = [&](int kt, int buf) {
    const int k0 = kt * BK;
    tdm_load_tile_bf16((unsigned)(uintptr_t)&sA[buf][0][0],
                       A + (size_t)bm * K + k0,
                       (unsigned)(K - k0), (unsigned)(M - bm),
                       BK, BM, (unsigned long long)K);
    tdm_load_tile_bf16((unsigned)(uintptr_t)&sB[buf][0][0],
                       W + (size_t)bn * K + k0,
                       (unsigned)(K - k0), (unsigned)(N - bn),
                       BK, BN, (unsigned long long)K);
  };

  auto compute = [&](int buf) {
    v16bf af[2];
    v16bf bfrag[4];
    // A fragment: 16-bit A 16x32 layout (lane<16: rows 0..15, K 0..7 & 16..23;
    // lane>=16: same rows, K 8..15 & 24..31; 2 K-values per VGPR)
#pragma unroll
    for (int mt = 0; mt < 2; ++mt) {
      const int row = wm * 32 + mt * 16 + ln;
#pragma unroll
      for (int v = 0; v < 8; ++v) {
        const int kb = ((v & 3) << 1) + (hi << 3) + ((v >> 2) << 4);
        af[mt][2 * v]     = sA[buf][row][kb];
        af[mt][2 * v + 1] = sA[buf][row][kb + 1];
      }
    }
    // B fragment: column N = lane%16; lanes 0-15 hold K 0..7 & 16..23, lanes 16-31 K 8..15 & 24..31
#pragma unroll
    for (int nt = 0; nt < 4; ++nt) {
      const int col = wn * 64 + nt * 16 + ln;
#pragma unroll
      for (int v = 0; v < 8; ++v) {
        const int k = ((v & 3) << 1) + (hi << 3) + ((v >> 2) << 4);
        bfrag[nt][2 * v]     = sB[buf][col][k];
        bfrag[nt][2 * v + 1] = sB[buf][col][k + 1];
      }
    }
#pragma unroll
    for (int mt = 0; mt < 2; ++mt)
#pragma unroll
      for (int nt = 0; nt < 4; ++nt)
        acc[mt][nt] = __builtin_amdgcn_wmma_f32_16x16x32_bf16(
            false, af[mt], false, bfrag[nt], (short)0, acc[mt][nt], false, false);
  };

  if (wave == 0) issue(0, 0);

  for (int kt = 0; kt < nk; ++kt) {
    const int cur = kt & 1;
    if (wave == 0) {
      if (kt + 1 < nk) {
        issue(kt + 1, cur ^ 1);                  // run DMA one tile ahead
        __builtin_amdgcn_s_wait_tensorcnt(2);    // current pair done (in-order)
      } else {
        __builtin_amdgcn_s_wait_tensorcnt(0);    // drain
      }
    }
    __syncthreads();       // publish TDM-written tile to all waves
    compute(cur);
    __syncthreads();       // everyone done reading before buffer reuse
  }

  // C/D layout: VGPR v, lanes 0-15 -> M=v, N=lane; lanes 16-31 -> M=v+8, N=lane-16
#pragma unroll
  for (int mt = 0; mt < 2; ++mt) {
#pragma unroll
    for (int v = 0; v < 8; ++v) {
      const int m = bm + wm * 32 + mt * 16 + hi * 8 + v;
#pragma unroll
      for (int nt = 0; nt < 4; ++nt) {
        const int n = bn + wn * 64 + nt * 16 + ln;
        float r = acc[mt][nt][v];
        if (bias) r += bias[n];
        O[(size_t)m * N + n] = r;
      }
    }
  }
}

// ---------------------------------------------------------------------------
// fp32 -> bf16 staging pass (done once per operand; halves GEMM HBM traffic)
// ---------------------------------------------------------------------------
__global__ __launch_bounds__(256) void cvt_f32_bf16_kernel(
    const float* __restrict__ in, bf16_t* __restrict__ out, size_t n4)
{
  const size_t i = ((size_t)blockIdx.x * blockDim.x + threadIdx.x);
  if (i >= n4) return;
  const float4 v = *(const float4*)(in + 4 * i);
  bf16_t* o = out + 4 * i;
  o[0] = f2bf(v.x); o[1] = f2bf(v.y); o[2] = f2bf(v.z); o[3] = f2bf(v.w);
}

// ---------------------------------------------------------------------------
// Causal depthwise conv (DC=4) over time + SiLU, with GQA-style channel repeat
// of x (DXB -> DIN): din channel c reads x channel ((c>>6)<<4) | (c&15).
// ---------------------------------------------------------------------------
__global__ __launch_bounds__(256) void conv_silu_kernel(
    const float* __restrict__ x, const float* __restrict__ conv_w,
    const float* __restrict__ conv_b, float* __restrict__ u)
{
  const size_t i = (size_t)blockIdx.x * blockDim.x + threadIdx.x;
  const size_t total = (size_t)B_SZ * L_SZ * D_IN;
  if (i >= total) return;
  const int c = (int)(i % D_IN);
  const size_t bt = i / D_IN;
  const int t = (int)(bt % L_SZ);
  const size_t b = bt / L_SZ;
  const int sc = ((c >> 6) << 4) | (c & 15);
  const float* w = conv_w + (size_t)c * D_C;
  float acc = conv_b[c];
#pragma unroll
  for (int j = 0; j < D_C; ++j) {
    const int tt = t - (D_C - 1) + j;
    if (tt >= 0) acc += w[j] * x[(b * L_SZ + tt) * (size_t)D_XB + sc];
  }
  u[i] = acc / (1.f + __expf(-acc));   // SiLU
}

// delta = softplus(dt_lin + bdt)   (dt_lin already carries one bdt from GEMM bias;
// reference adds the bias twice before softplus)
__global__ __launch_bounds__(256) void softplus_kernel(
    float* __restrict__ dtl, const float* __restrict__ bdt)
{
  const size_t i = (size_t)blockIdx.x * blockDim.x + threadIdx.x;
  const size_t total = (size_t)B_SZ * L_SZ * D_IN;
  if (i >= total) return;
  const int c = (int)(i % D_IN);
  const float v = dtl[i] + bdt[c];
  dtl[i] = (v > 20.f) ? v : log1pf(__expf(v));
}

// ---------------------------------------------------------------------------
// Selective scan. 16 lanes per din channel (ds=16 states, one per lane).
// Block = 256 threads = 16 channels; grid = (DIN/16, B). __shfl_xor reduces
// the 16 states; lane s==0 writes y[t]. 4096 waves in flight hide the
// L-serial load latency; 5 x 128MB streamed = ~27us at 23.3 TB/s.
// ---------------------------------------------------------------------------
__global__ __launch_bounds__(256) void scan_kernel(
    const float* __restrict__ u, const float* __restrict__ delta,
    const float* __restrict__ Bm, const float* __restrict__ Cm,
    const float* __restrict__ A_log, float* __restrict__ y)
{
  const int lane = threadIdx.x & 31;
  const int wave = threadIdx.x >> 5;
  const int s = lane & 15;
  const int c = blockIdx.x * 16 + wave * 2 + (lane >> 4);
  const int b = blockIdx.y;

  const float Ac = -__expf(A_log[(size_t)c * D_S + s]);
  const int bh = c >> 6;            // source B head (GQA repeat by 4)
  const int cbase = (c >> 4) << 4;  // C head base
  float h = 0.f;
  const size_t base = (size_t)b * L_SZ;

  for (int t = 0; t < L_SZ; ++t) {
    const size_t row = (base + t) * (size_t)D_IN;
    const float dt = delta[row + c];
    const float uv = u[row + c];
    const float Bv = Bm[(base + t) * (size_t)D_XB + (bh << 4) + s];
    const float Cv = Cm[row + cbase + s];
    h = __expf(dt * Ac) * h + (dt * uv) * Bv;
    float p = h * Cv;
    p += __shfl_xor(p, 1, 32);
    p += __shfl_xor(p, 2, 32);
    p += __shfl_xor(p, 4, 32);
    p += __shfl_xor(p, 8, 32);
    if (s == 0) y[row + c] = p;
  }
}

// y = (y + u*D) * silu(z)   (in place over y)
__global__ __launch_bounds__(256) void combine_kernel(
    float* __restrict__ y, const float* __restrict__ u,
    const float* __restrict__ z, const float* __restrict__ D)
{
  const size_t i = (size_t)blockIdx.x * blockDim.x + threadIdx.x;
  const size_t total = (size_t)B_SZ * L_SZ * D_IN;
  if (i >= total) return;
  const int c = (int)(i % D_IN);
  const float zv = z[i];
  const float sg = zv / (1.f + __expf(-zv));
  y[i] = (y[i] + u[i] * D[c]) * sg;
}

// ---------------------------------------------------------------------------
static void launch_gemm(const bf16_t* A, const bf16_t* W, const float* bias,
                        float* O, int M, int N, int K, hipStream_t stream)
{
  dim3 grid(N / BN, M / BM);
  gemm_bf16_tdm<<<grid, 256, 0, stream>>>(A, W, bias, O, M, N, K);
}

static void launch_cvt(const float* in, bf16_t* out, size_t n, hipStream_t stream)
{
  const size_t n4 = n / 4;                   // all sizes here are multiples of 4
  const int blk = 256;
  cvt_f32_bf16_kernel<<<(unsigned)((n4 + blk - 1) / blk), blk, 0, stream>>>(in, out, n4);
}

extern "C" void kernel_launch(void* const* d_in, const int* in_sizes, int n_in,
                              void* d_out, int out_size, void* d_ws, size_t ws_size,
                              hipStream_t stream) {
  const float* hs       = (const float*)d_in[0];   // (B,L,DM)
  const float* Wx       = (const float*)d_in[1];   // (DXB,DM)
  const float* bx       = (const float*)d_in[2];
  const float* Wz       = (const float*)d_in[3];   // (DIN,DM)
  const float* bz       = (const float*)d_in[4];
  const float* conv_w   = (const float*)d_in[5];   // (DIN,1,DC)
  const float* conv_b   = (const float*)d_in[6];
  const float* WB       = (const float*)d_in[7];   // (DXB,DM)
  const float* WC       = (const float*)d_in[8];   // (DIN,DM)
  const float* Wdt_down = (const float*)d_in[9];   // (DTR,DM)
  const float* Wdt      = (const float*)d_in[10];  // (DIN,DTR)
  const float* bdt      = (const float*)d_in[11];  // (DIN,)
  const float* A_log    = (const float*)d_in[12];  // (DIN,DS)
  const float* Dp       = (const float*)d_in[13];  // (DIN,)
  const float* Wout     = (const float*)d_in[14];  // (DM,DIN)
  const float* bout     = (const float*)d_in[15];  // (DM,)
  float* out            = (float*)d_out;           // (B,L,DM) fp32

  const int M = B_SZ * L_SZ;  // 8192

  // ---- carve scratch from d_ws: fp32 region first, bf16 region after ----
  float* x    = (float*)d_ws;                    // M x DXB
  float* z    = x    + (size_t)M * D_XB;         // M x DIN
  float* Bm   = z    + (size_t)M * D_IN;         // M x DXB
  float* Cm   = Bm   + (size_t)M * D_XB;         // M x DIN
  float* dtmp = Cm   + (size_t)M * D_IN;         // M x DTR
  float* dtl  = dtmp + (size_t)M * D_TR;         // M x DIN (dt_lin -> delta)
  float* u    = dtl  + (size_t)M * D_IN;         // M x DIN
  float* y    = u    + (size_t)M * D_IN;         // M x DIN

  bf16_t* hs_bf   = (bf16_t*)(y + (size_t)M * D_IN);
  bf16_t* Wx_bf   = hs_bf   + (size_t)M * D_M;
  bf16_t* Wz_bf   = Wx_bf   + (size_t)D_XB * D_M;
  bf16_t* WB_bf   = Wz_bf   + (size_t)D_IN * D_M;
  bf16_t* WC_bf   = WB_bf   + (size_t)D_XB * D_M;
  bf16_t* Wdtd_bf = WC_bf   + (size_t)D_IN * D_M;
  bf16_t* Wdt_bf  = Wdtd_bf + (size_t)D_TR * D_M;
  bf16_t* Wout_bf = Wdt_bf  + (size_t)D_IN * D_TR;
  bf16_t* dtmp_bf = Wout_bf + (size_t)D_M * D_IN;
  bf16_t* y_bf    = dtmp_bf + (size_t)M * D_TR;

  // ---- stage everything the GEMMs consume as bf16 (once) ----
  launch_cvt(hs,       hs_bf,   (size_t)M    * D_M,  stream);
  launch_cvt(Wx,       Wx_bf,   (size_t)D_XB * D_M,  stream);
  launch_cvt(Wz,       Wz_bf,   (size_t)D_IN * D_M,  stream);
  launch_cvt(WB,       WB_bf,   (size_t)D_XB * D_M,  stream);
  launch_cvt(WC,       WC_bf,   (size_t)D_IN * D_M,  stream);
  launch_cvt(Wdt_down, Wdtd_bf, (size_t)D_TR * D_M,  stream);
  launch_cvt(Wdt,      Wdt_bf,  (size_t)D_IN * D_TR, stream);
  launch_cvt(Wout,     Wout_bf, (size_t)D_M  * D_IN, stream);

  // ---- projections (TDM-fed bf16 WMMA GEMMs) ----
  launch_gemm(hs_bf,   Wx_bf,   bx,      x,    M, D_XB, D_M,  stream);
  launch_gemm(hs_bf,   Wz_bf,   bz,      z,    M, D_IN, D_M,  stream);
  launch_gemm(hs_bf,   WB_bf,   nullptr, Bm,   M, D_XB, D_M,  stream);
  launch_gemm(hs_bf,   WC_bf,   nullptr, Cm,   M, D_IN, D_M,  stream);
  launch_gemm(hs_bf,   Wdtd_bf, nullptr, dtmp, M, D_TR, D_M,  stream);
  launch_cvt(dtmp, dtmp_bf, (size_t)M * D_TR, stream);
  launch_gemm(dtmp_bf, Wdt_bf,  bdt,     dtl,  M, D_IN, D_TR, stream);

  // ---- elementwise + scan (fp32, bandwidth-bound) ----
  const size_t totIN = (size_t)M * D_IN;
  const int blk = 256;
  const int gIN = (int)((totIN + blk - 1) / blk);

  conv_silu_kernel<<<gIN, blk, 0, stream>>>(x, conv_w, conv_b, u);
  softplus_kernel <<<gIN, blk, 0, stream>>>(dtl, bdt);

  dim3 sgrid(D_IN / 16, B_SZ);
  scan_kernel<<<sgrid, 256, 0, stream>>>(u, dtl, Bm, Cm, A_log, y);

  combine_kernel<<<gIN, blk, 0, stream>>>(y, u, z, Dp);

  // ---- out = y @ Wout^T + bout ----
  launch_cvt(y, y_bf, (size_t)M * D_IN, stream);
  launch_gemm(y_bf, Wout_bf, bout, out, M, D_M, D_IN, stream);
}
